// TransformerBlock_25134148616691
// MI455X (gfx1250) — matrix operand order repaired and checked by reading, not compiled, not run on hardware
//
#include <hip/hip_runtime.h>
#include <math.h>

// ---------------------------------------------------------------------------
// Types for CDNA5 WMMA (gfx1250, wave32)
// ---------------------------------------------------------------------------
typedef __bf16 bf16;
typedef __attribute__((ext_vector_type(16))) __bf16 bf16x16;
typedef __attribute__((ext_vector_type(8)))  float  f32x8;
typedef unsigned int u32x4 __attribute__((ext_vector_type(4)));
typedef int          i32x8 __attribute__((ext_vector_type(8)));
typedef int          i32x4 __attribute__((ext_vector_type(4)));

union Frag16 { uint4 u[2]; bf16x16 v; };

#if defined(__AMDGCN__) && __has_builtin(__builtin_amdgcn_tensor_load_to_lds)
#define USE_TDM 1
#else
#define USE_TDM 0
#endif

// A-matrix 16x32 bf16 fragment (ISA 7.12.2):
// lanes 0-15: row M=lane, K = {0..7, 16..23}; lanes 16-31: K = {8..15, 24..31}
__device__ inline bf16x16 load_frag_a16(const bf16* base, int stride, int lane) {
  const int half = lane >> 4;
  const int r    = lane & 15;
  const bf16* p  = base + r * stride + half * 8;
  Frag16 f;
  f.u[0] = *(const uint4*)(p);
  f.u[1] = *(const uint4*)(p + 16);
  return f.v;
}

// B-matrix 32x16 bf16 fragment: lane = N col; lanes 0-15 K = {0..7, 16..23}; lanes 16-31: K = {8..15, 24..31}
__device__ inline bf16x16 load_frag_b16(const bf16* base, int stride, int lane) {
  const int half = lane >> 4;
  const int c    = lane & 15;
  const bf16* p  = base + c * stride + half * 8;
  Frag16 f;
  f.u[0] = *(const uint4*)(p);
  f.u[1] = *(const uint4*)(p + 16);
  return f.v;
}

__device__ inline f32x8 wmma_bf16(bf16x16 a, bf16x16 b, f32x8 c) {
  return __builtin_amdgcn_wmma_f32_16x16x32_bf16(false, a, false, b,
                                                 (short)0, c, false, false);
}

#if USE_TDM
typedef __attribute__((address_space(3))) char lds_char;
__device__ inline unsigned lds_addr_of(const void* p) {
  return (unsigned)(unsigned long long)(lds_char*)p;
}

// TDM 2D tile load: tile_dim0=32 (K, bf16), tile_dim1=128 rows, LDS row stride
// 40 shorts via pad_enable (pad 4 DWORDs every 16 DWORDs). ISA 8.3/8.4.
__device__ inline void tdm_load_tile(const bf16* gbase, unsigned lds_off,
                                     unsigned tdim0, unsigned tdim1,
                                     unsigned long long stride0) {
  const unsigned long long ga = (unsigned long long)(uintptr_t)gbase;
  u32x4 g0;
  g0[0] = 1u;                                        // count=1, user descriptor
  g0[1] = lds_off;                                   // lds_addr (bytes)
  g0[2] = (unsigned)(ga & 0xFFFFFFFFull);            // global_addr[31:0]
  g0[3] = (unsigned)((ga >> 32) & 0x01FFFFFFull)     // global_addr[56:32]
          | (2u << 30);                              // type = 2 ("image")
  i32x8 g1;
  g1[0] = (int)((1u << 16)      // data_size = 1 -> 2 bytes
              | (1u << 20)      // pad_enable
              | (3u << 22)      // pad_interval code 3 = 16 DWORDs (one 64B row)
              | (3u << 25));    // pad_amount  code 3 = 4 DWORDs (16B pad)
  g1[1] = (int)((tdim0 & 0xFFFFu) << 16);                        // tensor_dim0 lo
  g1[2] = (int)((tdim0 >> 16) | ((tdim1 & 0xFFFFu) << 16));      // dim0 hi|dim1 lo
  g1[3] = (int)((tdim1 >> 16) | (32u << 16));                    // dim1 hi|tile_dim0
  g1[4] = (int)128u;                                             // tile_dim1 (tile_dim2=0)
  g1[5] = (int)(unsigned)(stride0 & 0xFFFFFFFFull);              // dim0 stride lo
  g1[6] = (int)(unsigned)((stride0 >> 32) & 0xFFFFull);          // stride hi (dim1 stride=0)
  g1[7] = 0;
  const i32x4 z4 = {0, 0, 0, 0};
  const i32x8 z8 = {0, 0, 0, 0, 0, 0, 0, 0};
  // 6-arg form (clang-23 / therock-10.0 headers): groups 2/3 + extra group zero
  __builtin_amdgcn_tensor_load_to_lds(g0, g1, z4, z4, z8, 0);
}
#endif

// ---------------------------------------------------------------------------
// Problem dims
// ---------------------------------------------------------------------------
#define BB   2
#define TT   2048
#define DD   1024
#define HH   16
#define HD   64
#define DFF  4096
#define BT   (BB * TT)

// ---------------------------------------------------------------------------
// Weight prep: W f32 [K,N] -> Wt bf16 [N,K] (one-time, ~2us of HBM traffic)
// ---------------------------------------------------------------------------
__global__ __launch_bounds__(256) void wt_kernel(const float* __restrict__ W,
                                                 bf16* __restrict__ Wt,
                                                 int K, int N) {
  __shared__ float tile[32][33];
  const int n0 = blockIdx.x * 32;
  const int k0 = blockIdx.y * 32;
  const int tx = threadIdx.x & 31;
  const int ty = threadIdx.x >> 5;        // 8 rows per pass
#pragma unroll
  for (int i = 0; i < 32; i += 8)
    tile[ty + i][tx] = W[(size_t)(k0 + ty + i) * N + n0 + tx];
  __syncthreads();
#pragma unroll
  for (int i = 0; i < 32; i += 8)
    Wt[(size_t)(n0 + ty + i) * K + k0 + tx] = (bf16)tile[tx][ty + i];
}

// ---------------------------------------------------------------------------
// LayerNorm: fp32 in -> bf16 out (one block per row)
// ---------------------------------------------------------------------------
__global__ __launch_bounds__(256) void ln_kernel(const float* __restrict__ x,
                                                 const float* __restrict__ scale,
                                                 const float* __restrict__ shift,
                                                 bf16* __restrict__ out) {
  const int row = blockIdx.x;
  const float* xr = x + (size_t)row * DD;
  float s = 0.f, s2 = 0.f;
  for (int c = threadIdx.x; c < DD; c += 256) {
    const float v = xr[c];
    s += v; s2 += v * v;
  }
  __shared__ float r1[256], r2[256];
  r1[threadIdx.x] = s; r2[threadIdx.x] = s2;
  __syncthreads();
  for (int off = 128; off > 0; off >>= 1) {
    if (threadIdx.x < off) {
      r1[threadIdx.x] += r1[threadIdx.x + off];
      r2[threadIdx.x] += r2[threadIdx.x + off];
    }
    __syncthreads();
  }
  const float mean = r1[0] * (1.0f / DD);
  const float var  = r2[0] * (1.0f / DD) - mean * mean;
  const float rstd = rsqrtf(var + 1e-5f);
  bf16* orow = out + (size_t)row * DD;
  for (int c = threadIdx.x; c < DD; c += 256)
    orow[c] = (bf16)(((xr[c] - mean) * rstd) * scale[c] + shift[c]);
}

// ---------------------------------------------------------------------------
// GEMM: out = act(A_bf16[M,K] @ Wt_bf16[N,K]^T + bias) (+ resid)
// 128x128x32 tile, 8 waves (4Mx2N), wave tile 32x64 -> 8 WMMA accumulators.
// Steady-state loop: TDM double-buffered tile staging + ds_load_b128 + v_wmma.
// ---------------------------------------------------------------------------
__global__ __launch_bounds__(256) void gemm_kernel(
    const bf16*  __restrict__ A, const bf16* __restrict__ Wt,
    const float* __restrict__ bias, const float* __restrict__ resid,
    float* __restrict__ outF, bf16* __restrict__ outB,
    int M, int N, int K, int act)
{
  __shared__ __align__(16) bf16 As[2][128][40];   // [buf][m][k], 40 = 16B-aligned pad
  __shared__ __align__(16) bf16 Bs[2][128][40];   // [buf][n][k]
  const int tid  = threadIdx.x;
  const int lane = tid & 31;
  const int wid  = tid >> 5;
  const int half = lane >> 4;
  const int ln   = lane & 15;
  const int m0 = blockIdx.y * 128;
  const int n0 = blockIdx.x * 128;
  const int wm = wid & 3;            // 32-row group
  const int wn = wid >> 2;           // 64-col group

  const f32x8 zf = {0.f, 0.f, 0.f, 0.f, 0.f, 0.f, 0.f, 0.f};
  f32x8 acc[2][4];
#pragma unroll
  for (int i = 0; i < 2; i++)
#pragma unroll
    for (int j = 0; j < 4; j++) acc[i][j] = zf;

#if USE_TDM
  // wave 0 drives the Tensor Data Mover; everyone else just computes
  if (wid == 0) {
    tdm_load_tile(A  + (size_t)m0 * K, lds_addr_of(&As[0][0][0]), K, M - m0, K);
    tdm_load_tile(Wt + (size_t)n0 * K, lds_addr_of(&Bs[0][0][0]), K, N - n0, K);
  }
  for (int k0 = 0, it = 0; k0 < K; k0 += 32, ++it) {
    const int buf = it & 1;
    if (wid == 0) asm volatile("s_wait_tensorcnt 0x0" ::: "memory");
    __syncthreads();
    if (wid == 0 && k0 + 32 < K) {
      tdm_load_tile(A  + (size_t)m0 * K + (k0 + 32),
                    lds_addr_of(&As[buf ^ 1][0][0]), K - k0 - 32, M - m0, K);
      tdm_load_tile(Wt + (size_t)n0 * K + (k0 + 32),
                    lds_addr_of(&Bs[buf ^ 1][0][0]), K - k0 - 32, N - n0, K);
    }
#else
  const int rowS = tid >> 1, segS = (tid & 1) * 16;   // 2 thr/row, 16 elems each
  for (int k0 = 0, it = 0; k0 < K; k0 += 32, ++it) {
    const int buf = it & 1;
    {
      const uint4* ga = (const uint4*)(A + (size_t)(m0 + rowS) * K + k0 + segS);
      const uint4 a0 = ga[0], a1 = ga[1];
      *(uint4*)(&As[buf][rowS][segS])     = a0;
      *(uint4*)(&As[buf][rowS][segS + 8]) = a1;
      const uint4* gb = (const uint4*)(Wt + (size_t)(n0 + rowS) * K + k0 + segS);
      const uint4 b0 = gb[0], b1 = gb[1];
      *(uint4*)(&Bs[buf][rowS][segS])     = b0;
      *(uint4*)(&Bs[buf][rowS][segS + 8]) = b1;
    }
    __syncthreads();
#endif
    bf16x16 af[2], bw[4];
#pragma unroll
    for (int ms = 0; ms < 2; ms++)
      af[ms] = load_frag_a16(&As[buf][wm * 32 + ms * 16][0], 40, lane);
#pragma unroll
    for (int ns = 0; ns < 4; ns++)
      bw[ns] = load_frag_b16(&Bs[buf][wn * 64 + ns * 16][0], 40, lane);
#pragma unroll
    for (int ms = 0; ms < 2; ms++)
#pragma unroll
      for (int ns = 0; ns < 4; ns++)
        acc[ms][ns] = wmma_bf16(af[ms], bw[ns], acc[ms][ns]);
    __syncthreads();
  }

  // fused epilogue; C layout: VGPR r -> row r + 8*half, lane -> col
  const bool hasBias = (bias != nullptr);
  const bool hasRes  = (resid != nullptr);
#pragma unroll
  for (int ms = 0; ms < 2; ms++) {
#pragma unroll
    for (int ns = 0; ns < 4; ns++) {
      const f32x8 c = acc[ms][ns];
#pragma unroll
      for (int r = 0; r < 8; r++) {
        const int row = m0 + wm * 32 + ms * 16 + r + half * 8;
        const int col = n0 + wn * 64 + ns * 16 + ln;
        float vv = c[r];
        if (hasBias) vv += bias[col];
        if (act) {
          // 0.5u(1+tanh(t)) == u * sigmoid(2t): single v_exp_f32, no libm branches
          const float u = vv;
          const float t = 0.7978845608028654f * (u + 0.044715f * u * u * u);
          vv = u / (1.0f + __expf(-2.0f * t));
        }
        const size_t idx = (size_t)row * N + col;
        if (hasRes) vv += resid[idx];
        if (outF) outF[idx] = vv;
        if (outB) outB[idx] = (bf16)vv;
      }
    }
  }
}

// ---------------------------------------------------------------------------
// Causal flash attention: one wave per (b, h, 16-row q tile); waves independent.
// q,k,v,ctx are bf16 [B*T, D] with head offset h*64.
// ---------------------------------------------------------------------------
__global__ __launch_bounds__(256) void attn_kernel(const bf16* __restrict__ q,
                                                   const bf16* __restrict__ k,
                                                   const bf16* __restrict__ v,
                                                   bf16* __restrict__ ctx) {
  __shared__ __align__(16) bf16 Pbuf[8][16][40];   // per-wave P tile 16x32
  __shared__ __align__(16) bf16 Vbuf[8][64][40];   // per-wave V^T tile [hd][t]
  const int tid  = threadIdx.x;
  const int lane = tid & 31;
  const int wid  = tid >> 5;
  const int half = lane >> 4;
  const int ln   = lane & 15;

  const int task = blockIdx.x * 8 + wid;   // B*H*(T/16) = 4096 tasks
  const int qt = task & 127;               // T/16 = 128
  const int bh = task >> 7;
  const int b  = bh >> 4;
  const int h  = bh & 15;
  const int q0 = qt * 16;
  const size_t baseRow = (size_t)b * TT;
  const size_t hoff    = (size_t)h * HD;

  // Q fragments (head dim 64 -> two 16x32 A fragments) straight from global
  bf16x16 qa[2];
  {
    const size_t rowoff = (baseRow + q0 + ln) * DD + hoff;
#pragma unroll
    for (int f = 0; f < 2; f++) {
      const bf16* p = q + rowoff + f * 32 + half * 8;
      Frag16 fr;
      fr.u[0] = *(const uint4*)(p);
      fr.u[1] = *(const uint4*)(p + 16);
      qa[f] = fr.v;
    }
  }

  const f32x8 zf = {0.f, 0.f, 0.f, 0.f, 0.f, 0.f, 0.f, 0.f};
  float mrow[8], lrow[8];
  f32x8 o[4];
#pragma unroll
  for (int r = 0; r < 8; r++) { mrow[r] = -3.0e38f; lrow[r] = 0.f; }
#pragma unroll
  for (int t = 0; t < 4; t++) o[t] = zf;

  const int kend = q0 + 15;
  for (int kt = 0; kt <= kend; kt += 32) {
    // ---- S = (Q K^T) * 1/sqrt(64); K^T B-fragments contiguous in hd ----
    f32x8 s[2];
#pragma unroll
    for (int nt = 0; nt < 2; nt++) {
      const int tcol = kt + nt * 16 + ln;
      f32x8 z = zf;
#pragma unroll
      for (int f = 0; f < 2; f++) {
        const bf16* p = k + (baseRow + tcol) * DD + hoff + f * 32 + half * 8;
        Frag16 fr;
        fr.u[0] = *(const uint4*)(p);
        fr.u[1] = *(const uint4*)(p + 16);
        z = wmma_bf16(qa[f], fr.v, z);
      }
      s[nt] = z;
    }
    // mask + scale
    float sv[2][8];
#pragma unroll
    for (int nt = 0; nt < 2; nt++) {
      const int tcol = kt + nt * 16 + ln;
#pragma unroll
      for (int r = 0; r < 8; r++) {
        const int qrow = q0 + r + half * 8;
        const float xx = s[nt][r] * 0.125f;
        sv[nt][r] = (tcol <= qrow) ? xx : -3.0e38f;
      }
    }
    // online softmax: reduce across the 16 lanes sharing each row
    float alpha[8], mnew[8];
#pragma unroll
    for (int r = 0; r < 8; r++) {
      float tm = fmaxf(sv[0][r], sv[1][r]);
      tm = fmaxf(tm, __shfl_xor(tm, 1, 32));
      tm = fmaxf(tm, __shfl_xor(tm, 2, 32));
      tm = fmaxf(tm, __shfl_xor(tm, 4, 32));
      tm = fmaxf(tm, __shfl_xor(tm, 8, 32));
      mnew[r]  = fmaxf(mrow[r], tm);
      alpha[r] = __expf(mrow[r] - mnew[r]);
      mrow[r]  = mnew[r];
    }
    float psum[8];
#pragma unroll
    for (int r = 0; r < 8; r++) psum[r] = 0.f;
#pragma unroll
    for (int nt = 0; nt < 2; nt++) {
#pragma unroll
      for (int r = 0; r < 8; r++) {
        const float p = __expf(sv[nt][r] - mnew[r]);
        psum[r] += p;
        Pbuf[wid][r + half * 8][nt * 16 + ln] = (bf16)p;
      }
    }
#pragma unroll
    for (int r = 0; r < 8; r++) {
      float ps = psum[r];
      ps += __shfl_xor(ps, 1, 32);
      ps += __shfl_xor(ps, 2, 32);
      ps += __shfl_xor(ps, 4, 32);
      ps += __shfl_xor(ps, 8, 32);
      lrow[r] = lrow[r] * alpha[r] + ps;
    }
    // rescale running output
#pragma unroll
    for (int t = 0; t < 4; t++)
#pragma unroll
      for (int r = 0; r < 8; r++) o[t][r] *= alpha[r];
    // stage V tile transposed into per-wave LDS: [hd][t_local]
    {
      const bf16* vs = v + (baseRow + kt + lane) * DD + hoff;
#pragma unroll
      for (int hd = 0; hd < HD; hd++) Vbuf[wid][hd][lane] = vs[hd];
    }
    asm volatile("s_wait_dscnt 0x0" ::: "memory");   // same-wave LDS W->R
    // O += P @ V
    const bf16x16 pa = load_frag_a16(&Pbuf[wid][0][0], 40, lane);
#pragma unroll
    for (int t = 0; t < 4; t++) {
      const bf16x16 vb = load_frag_b16(&Vbuf[wid][t * 16][0], 40, lane);
      o[t] = wmma_bf16(pa, vb, o[t]);
    }
    asm volatile("s_wait_dscnt 0x0" ::: "memory");
  }

  // normalize + store ctx (bf16)
  float rl[8];
#pragma unroll
  for (int r = 0; r < 8; r++) rl[r] = 1.0f / lrow[r];
#pragma unroll
  for (int t = 0; t < 4; t++) {
#pragma unroll
    for (int r = 0; r < 8; r++) {
      const int row = q0 + r + half * 8;
      const int col = t * 16 + ln;
      ctx[(baseRow + row) * DD + hoff + col] = (bf16)(o[t][r] * rl[r]);
    }
  }
}

// ---------------------------------------------------------------------------
// Host-side orchestration
// ---------------------------------------------------------------------------
#define MB(x) ((size_t)(x) << 20)

extern "C" void kernel_launch(void* const* d_in, const int* in_sizes, int n_in,
                              void* d_out, int out_size, void* d_ws, size_t ws_size,
                              hipStream_t stream) {
  (void)in_sizes; (void)n_in; (void)out_size; (void)ws_size;
  const float* x    = (const float*)d_in[0];
  const float* wq   = (const float*)d_in[1];
  const float* wk   = (const float*)d_in[2];
  const float* wv   = (const float*)d_in[3];
  const float* wo   = (const float*)d_in[4];
  const float* bo   = (const float*)d_in[5];
  const float* ln1s = (const float*)d_in[6];
  const float* ln1b = (const float*)d_in[7];
  const float* ln2s = (const float*)d_in[8];
  const float* ln2b = (const float*)d_in[9];
  const float* w1   = (const float*)d_in[10];
  const float* b1   = (const float*)d_in[11];
  const float* w2   = (const float*)d_in[12];
  const float* b2   = (const float*)d_in[13];
  float* out = (float*)d_out;

  char* ws = (char*)d_ws;
  bf16*  h1  = (bf16*)(ws + MB(0));     //  8 MiB [BT,D] bf16
  bf16*  qb  = (bf16*)(ws + MB(8));     //  8 MiB
  bf16*  kb  = (bf16*)(ws + MB(16));    //  8 MiB
  bf16*  vb  = (bf16*)(ws + MB(24));    //  8 MiB
  bf16*  cb  = (bf16*)(ws + MB(32));    //  8 MiB ctx
  float* x1  = (float*)(ws + MB(40));   // 16 MiB residual 1
  bf16*  h2  = (bf16*)(ws + MB(56));    //  8 MiB
  bf16*  ffa = (bf16*)(ws + MB(64));    // 32 MiB [BT,DFF] bf16
  bf16*  wqT = (bf16*)(ws + MB(96));    //  2 MiB [N,K] bf16
  bf16*  wkT = (bf16*)(ws + MB(98));
  bf16*  wvT = (bf16*)(ws + MB(100));
  bf16*  woT = (bf16*)(ws + MB(102));
  bf16*  w1T = (bf16*)(ws + MB(104));   //  8 MiB [DFF,D]
  bf16*  w2T = (bf16*)(ws + MB(112));   //  8 MiB [D,DFF]  -> total 120 MiB

  const dim3 blk(256);
  const dim3 gD (DD  / 128, BT / 128);  // (8, 32)
  const dim3 gF (DFF / 128, BT / 128);  // (32, 32)

  // 0. one-time weight convert+transpose to bf16 [N,K]
  wt_kernel<<<dim3(DD  / 32, DD  / 32), blk, 0, stream>>>(wq, wqT, DD, DD);
  wt_kernel<<<dim3(DD  / 32, DD  / 32), blk, 0, stream>>>(wk, wkT, DD, DD);
  wt_kernel<<<dim3(DD  / 32, DD  / 32), blk, 0, stream>>>(wv, wvT, DD, DD);
  wt_kernel<<<dim3(DD  / 32, DD  / 32), blk, 0, stream>>>(wo, woT, DD, DD);
  wt_kernel<<<dim3(DFF / 32, DD  / 32), blk, 0, stream>>>(w1, w1T, DD, DFF);
  wt_kernel<<<dim3(DD  / 32, DFF / 32), blk, 0, stream>>>(w2, w2T, DFF, DD);

  // 1. h = LN1(x)
  ln_kernel<<<BT, blk, 0, stream>>>(x, ln1s, ln1b, h1);
  // 2-4. q,k,v = h @ W{q,k,v}
  gemm_kernel<<<gD, blk, 0, stream>>>(h1, wqT, nullptr, nullptr, nullptr, qb, BT, DD, DD, 0);
  gemm_kernel<<<gD, blk, 0, stream>>>(h1, wkT, nullptr, nullptr, nullptr, kb, BT, DD, DD, 0);
  gemm_kernel<<<gD, blk, 0, stream>>>(h1, wvT, nullptr, nullptr, nullptr, vb, BT, DD, DD, 0);
  // 5. ctx = causal softmax(q k^T / 8) v
  attn_kernel<<<BB * HH * (TT / 16) / 8, blk, 0, stream>>>(qb, kb, vb, cb);
  // 6. x1 = x + ctx @ wo + bo
  gemm_kernel<<<gD, blk, 0, stream>>>(cb, woT, bo, x, x1, nullptr, BT, DD, DD, 0);
  // 7. h2 = LN2(x1)
  ln_kernel<<<BT, blk, 0, stream>>>(x1, ln2s, ln2b, h2);
  // 8. ffa = gelu(h2 @ w1 + b1)
  gemm_kernel<<<gF, blk, 0, stream>>>(h2, w1T, b1, nullptr, nullptr, ffa, BT, DFF, DD, 1);
  // 9. out = x1 + ffa @ w2 + b2
  gemm_kernel<<<gD, blk, 0, stream>>>(ffa, w2T, b2, x1, out, nullptr, BT, DD, DFF, 0);
}
